// CausalTrajectoryPrediction_4707284157197
// MI455X (gfx1250) — matrix operand order repaired and checked by reading, not compiled, not run on hardware
//
#include <hip/hip_runtime.h>

typedef __attribute__((ext_vector_type(16))) _Float16 v16h;
typedef __attribute__((ext_vector_type(8)))  float    v8f;

#define LDH 72  // padded leading dim (halves) for 16x64 activation tiles

__device__ __forceinline__ v8f wmma_f16(v16h a, v16h b, v8f c) {
  // D = A(16x32 f16) * B(32x16 f16) + C(16x16 f32)
  return __builtin_amdgcn_wmma_f32_16x16x32_f16(false, a, false, b, (short)0, c,
                                                false, false);
}

// ---- block-cooperative weight staging: f32 global -> f16 LDS (row-major) ----
__device__ __forceinline__ void stage_w(const float* __restrict__ src,
                                        _Float16* __restrict__ dst, int nelem,
                                        int tid) {
  for (int t = tid * 4; t < nelem; t += 256 * 4) {
    const float4 v = *(const float4*)(src + t);
    dst[t + 0] = (_Float16)v.x;
    dst[t + 1] = (_Float16)v.y;
    dst[t + 2] = (_Float16)v.z;
    dst[t + 3] = (_Float16)v.w;
  }
}

// W3a: copy columns 0..63 of a [64][128] f32 matrix into a [64][64] f16 tile
__device__ __forceinline__ void stage_w3(const float* __restrict__ src,
                                         _Float16* __restrict__ dst, int tid) {
  for (int t = tid * 4; t < 4096; t += 256 * 4) {
    const int h = t >> 6, c = t & 63;
    const float4 v = *(const float4*)(src + h * 128 + c);
    _Float16* d = dst + t;
    d[0] = (_Float16)v.x;
    d[1] = (_Float16)v.y;
    d[2] = (_Float16)v.z;
    d[3] = (_Float16)v.w;
  }
}

// B fragment (32x16, K x N) from f16 LDS: lane = column (lane&15),
// lanes 0-15: K+{0..7,16..23}; lanes 16-31: K+{8..15,24..31} -> 2x ds_load_b128 per lane
__device__ __forceinline__ v16h load_bfrag_lds(const _Float16* __restrict__ W,
                                               int ld, int nbase, int kbase,
                                               int lane) {
  const int col = lane & 15;
  const int ko  = kbase + ((lane >> 4) << 3);
  const _Float16* p = W + (nbase + col) * ld + ko;
  v16h r;
#pragma unroll
  for (int e = 0; e < 8; ++e)  r[e] = p[e];
#pragma unroll
  for (int e = 8; e < 16; ++e) r[e] = p[e + 8];
  return r;
}

// A fragment (16x32, M x K) from f32 global, row-major ld=64.
// lanes 0-15: row=lane, K = koff+{0..7,16..23}; lanes 16-31: K = koff+{8..15,24..31}
__device__ __forceinline__ v16h load_afrag_x(const float* __restrict__ x, int b0,
                                             int koff, int lane) {
  const int row = lane & 15;
  const int kb  = (lane & 16) ? 8 : 0;
  const float* p = x + (size_t)(b0 + row) * 64 + koff + kb;
  v16h r;
#pragma unroll
  for (int e = 0; e < 8; ++e)  r[e] = (_Float16)p[e];
#pragma unroll
  for (int e = 8; e < 16; ++e) r[e] = (_Float16)p[e + 8];
  return r;
}

// Same A layout, sourced from an f16 LDS tile [16][LDH]
__device__ __forceinline__ v16h load_afrag_lds(const _Float16* __restrict__ t,
                                               int koff, int lane) {
  const int row = lane & 15;
  const int kb  = (lane & 16) ? 8 : 0;
  const _Float16* p = t + row * LDH + koff + kb;
  v16h r;
#pragma unroll
  for (int e = 0; e < 8; ++e)  r[e] = p[e];
#pragma unroll
  for (int e = 8; e < 16; ++e) r[e] = p[e + 8];
  return r;
}

// Store C tile (16x16 accumulator layout) as f16 into LDS tile at column nbase.
__device__ __forceinline__ void store_c_f16(_Float16* __restrict__ t, int nbase,
                                            const float* vals, int lane) {
  const int n  = lane & 15;
  const int mo = (lane & 16) ? 8 : 0;
#pragma unroll
  for (int r = 0; r < 8; ++r) t[(mo + r) * LDH + nbase + n] = (_Float16)vals[r];
}

__global__ __launch_bounds__(256) void ctp_wmma_kernel(
    const float* __restrict__ x,
    const float* __restrict__ W1a, const float* __restrict__ W1b,
    const float* __restrict__ W2a, const float* __restrict__ W2b,
    const float* __restrict__ W3a, const float* __restrict__ b3a,
    const float* __restrict__ W3b, const float* __restrict__ b3b,
    float* __restrict__ out) {
  __shared__ _Float16 sWa[64 * 64];     // branch layer-1 weights (reused br=0,1)
  __shared__ _Float16 sWb[32 * 64];     // branch layer-2 weights (reused br=0,1)
  __shared__ _Float16 sW3[64 * 64];     // fuse weights, feat cols 0..63
  __shared__ _Float16 sH[8][16 * LDH];  // per-wave hidden tile (16 x 64)
  __shared__ _Float16 sR[8][16 * LDH];  // per-wave [r1|r2] tile (16 x 64)

  const int tid  = threadIdx.x;
  const int lane = tid & 31;
  const int wave = tid >> 5;
  const int i    = blockIdx.y;                    // node index, 0..63
  const int b0   = (blockIdx.x * 8 + wave) * 16;  // batch tile base
  const int mo   = (lane & 16) ? 8 : 0;
  const int hn   = lane & 15;

  _Float16* hT = sH[wave];
  _Float16* rT = sR[wave];

  const float* W3g = W3a + (size_t)i * 64 * 128;  // [H=64, 2M+N=128]
  __builtin_prefetch(W3g, 0, 1);                  // global_prefetch_b8

  // stage fuse weights once (consumed after the branch-loop barriers)
  stage_w3(W3g, sW3, tid);

  // x A-fragments (K = 0..31 and 32..63), f32 -> f16
  v16h ax0 = load_afrag_x(x, b0, 0, lane);
  v16h ax1 = load_afrag_x(x, b0, 32, lane);

  // x[b, i] per accumulator row (rank-1 mask correction + x2 diag term)
  float xi[8];
#pragma unroll
  for (int r = 0; r < 8; ++r) xi[r] = x[(size_t)(b0 + mo + r) * 64 + i];

  // ---- two branches: (W1a,W1b) directed, (W2a,W2b) bi-directed ----
#pragma unroll 1
  for (int br = 0; br < 2; ++br) {
    const float* Wag = (br ? W2a : W1a) + (size_t)i * 64 * 64;  // [H=64, N=64]
    const float* Wbg = (br ? W2b : W1b) + (size_t)i * 32 * 64;  // [M=32, H=64]

    __syncthreads();  // previous consumers of sWa/sWb done
    stage_w(Wag, sWa, 64 * 64, tid);
    stage_w(Wbg, sWb, 32 * 64, tid);
    __syncthreads();  // staged weights visible

    // layer 1: h = relu(x @ Wa^T - x[b,i] * Wa[:,i])  -> hT (per-wave)
    // load-all-B, WMMA burst (k-interleaved), then epilogue
    {
      v16h bw[8];
#pragma unroll
      for (int g = 0; g < 4; ++g) {
        bw[2 * g]     = load_bfrag_lds(sWa, 64, g * 16, 0, lane);
        bw[2 * g + 1] = load_bfrag_lds(sWa, 64, g * 16, 32, lane);
      }
      float dg[4];
#pragma unroll
      for (int g = 0; g < 4; ++g) dg[g] = Wag[(size_t)(g * 16 + hn) * 64 + i];

      v8f c[4] = {};
#pragma unroll
      for (int g = 0; g < 4; ++g) c[g] = wmma_f16(ax0, bw[2 * g], c[g]);
#pragma unroll
      for (int g = 0; g < 4; ++g) c[g] = wmma_f16(ax1, bw[2 * g + 1], c[g]);

#pragma unroll
      for (int g = 0; g < 4; ++g) {
        float v[8];
#pragma unroll
        for (int r = 0; r < 8; ++r) v[r] = fmaxf(c[g][r] - xi[r] * dg[g], 0.f);
        store_c_f16(hT, g * 16, v, lane);
      }
    }
    // per-wave tile: LDS ops from one wave are in-order; no block barrier needed
    v16h ah0 = load_afrag_lds(hT, 0, lane);
    v16h ah1 = load_afrag_lds(hT, 32, lane);

    // layer 2: r = relu(h @ Wb^T)  -> rT columns [br*32, br*32+32)
    {
      v16h bw[4];
#pragma unroll
      for (int g = 0; g < 2; ++g) {
        bw[2 * g]     = load_bfrag_lds(sWb, 64, g * 16, 0, lane);
        bw[2 * g + 1] = load_bfrag_lds(sWb, 64, g * 16, 32, lane);
      }
      v8f c[2] = {};
#pragma unroll
      for (int g = 0; g < 2; ++g) c[g] = wmma_f16(ah0, bw[2 * g], c[g]);
#pragma unroll
      for (int g = 0; g < 2; ++g) c[g] = wmma_f16(ah1, bw[2 * g + 1], c[g]);

#pragma unroll
      for (int g = 0; g < 2; ++g) {
        float v[8];
#pragma unroll
        for (int r = 0; r < 8; ++r) v[r] = fmaxf(c[g][r], 0.f);
        store_c_f16(rT, br * 32 + g * 16, v, lane);
      }
    }
  }
  __syncthreads();  // sW3 staged long ago; rT per-wave complete

  // ---- fuse layer: h = relu(r1 @ W3a[:,0:32]^T + r2 @ W3a[:,32:64]^T
  //                           + x[b,i]*W3a[:,64+i] + b3a) ----
  v16h ar0 = load_afrag_lds(rT, 0, lane);   // r1, feat K=0..31
  v16h ar1 = load_afrag_lds(rT, 32, lane);  // r2, feat K=32..63

  float part[8];
#pragma unroll
  for (int r = 0; r < 8; ++r) part[r] = 0.f;

  {
    v16h bf[8];
#pragma unroll
    for (int g = 0; g < 4; ++g) {
      bf[2 * g]     = load_bfrag_lds(sW3, 64, g * 16, 0, lane);
      bf[2 * g + 1] = load_bfrag_lds(sW3, 64, g * 16, 32, lane);
    }
    float d3[4], ba[4], w3[4];
#pragma unroll
    for (int g = 0; g < 4; ++g) {
      const int h = g * 16 + hn;
      d3[g] = W3g[(size_t)h * 128 + 64 + i];  // x2 diag contribution
      ba[g] = b3a[i * 64 + h];
      w3[g] = W3b[i * 64 + h];                // output layer weight
    }
    v8f c[4] = {};
#pragma unroll
    for (int g = 0; g < 4; ++g) c[g] = wmma_f16(ar0, bf[2 * g], c[g]);
#pragma unroll
    for (int g = 0; g < 4; ++g) c[g] = wmma_f16(ar1, bf[2 * g + 1], c[g]);

#pragma unroll
    for (int g = 0; g < 4; ++g)
#pragma unroll
      for (int r = 0; r < 8; ++r) {
        const float hv = fmaxf(c[g][r] + xi[r] * d3[g] + ba[g], 0.f);
        part[r] += hv * w3[g];  // fold output dot-product immediately
      }
  }

  // ---- output: out[b, i] = relu(sum_h h*W3b + b3b); reduce over N lanes ----
  const float bb = b3b[i];
#pragma unroll
  for (int r = 0; r < 8; ++r) {
    float s = part[r];
    s += __shfl_xor(s, 1, 32);
    s += __shfl_xor(s, 2, 32);
    s += __shfl_xor(s, 4, 32);
    s += __shfl_xor(s, 8, 32);  // lanes 0 / 16 hold full sums for m=r / m=r+8
    if ((lane & 15) == 0)
      out[(size_t)(b0 + mo + r) * 64 + i] = fmaxf(s + bb, 0.f);
  }
}

extern "C" void kernel_launch(void* const* d_in, const int* in_sizes, int n_in,
                              void* d_out, int out_size, void* d_ws, size_t ws_size,
                              hipStream_t stream) {
  const float* x   = (const float*)d_in[0];
  const float* W1a = (const float*)d_in[1];
  const float* W1b = (const float*)d_in[2];
  const float* W2a = (const float*)d_in[3];
  const float* W2b = (const float*)d_in[4];
  const float* W3a = (const float*)d_in[5];
  const float* b3a = (const float*)d_in[6];
  const float* W3b = (const float*)d_in[7];
  const float* b3b = (const float*)d_in[8];
  float* out = (float*)d_out;

  const int B = in_sizes[0] / 64;  // 16384
  dim3 grid(B / 128, 64);          // 8 waves/block * 16 rows = 128 batch rows/block
  ctp_wmma_kernel<<<grid, 256, 0, stream>>>(x, W1a, W1b, W2a, W2b, W3a, b3a, W3b,
                                            b3b, out);
}
